// Upsample2d_63702954934467
// MI455X (gfx1250) — matrix operand order repaired and checked by reading, not compiled, not run on hardware
//
#include <hip/hip_runtime.h>
#include <hip/hip_bf16.h>

// ---------------------------------------------------------------------------
// Types for CDNA5 WMMA (wave32): v_wmma_f32_16x16x32_bf16
// ---------------------------------------------------------------------------
typedef __attribute__((ext_vector_type(16))) __bf16 v16bf;
typedef __attribute__((ext_vector_type(8)))  float  v8f;
typedef int v4i_t __attribute__((vector_size(16)));   // matches builtin's V4i

union Frag {
    v16bf v;
    uint4 u[2];
};

__device__ inline v8f zero8() {
    v8f z = {0.f, 0.f, 0.f, 0.f, 0.f, 0.f, 0.f, 0.f};
    return z;
}

// ---------------------------------------------------------------------------
// CDNA5 async global->LDS staging (ASYNCcnt path), with sync fallback.
// Builtin signature (probed from diagnostic): (v4i AS1*, v4i AS3*, Imm, Imm).
// ---------------------------------------------------------------------------
#if __has_builtin(__builtin_amdgcn_global_load_async_to_lds_b128)
#define HAVE_ASYNC_LDS 1
#endif

#ifdef HAVE_ASYNC_LDS
#define ASYNC_B128(gp, lp)                                                  \
    __builtin_amdgcn_global_load_async_to_lds_b128(                         \
        (__attribute__((address_space(1))) v4i_t*)(gp),                     \
        (__attribute__((address_space(3))) v4i_t*)(lp), 0, 0)
#endif

__device__ inline void wait_async0() {
#ifdef HAVE_ASYNC_LDS
#if __has_builtin(__builtin_amdgcn_s_wait_asynccnt)
    __builtin_amdgcn_s_wait_asynccnt(0);
#else
    asm volatile("s_wait_asynccnt 0" ::: "memory");
#endif
#endif
}

// Stage one 128x32 bf16 tile (row stride = strideElems) into LDS.
// Each of 256 threads moves 32 bytes: wrow = tid>>1 (row), whalf = tid&1.
__device__ inline void stage_w_tile(const __bf16* __restrict__ gtile,
                                    __bf16* ltile, int wrow, int whalf,
                                    size_t strideElems) {
    const __bf16* g = gtile + (size_t)wrow * strideElems + whalf * 16;
    __bf16* l = ltile + wrow * 32 + whalf * 16;
#ifdef HAVE_ASYNC_LDS
    ASYNC_B128(g, l);
    ASYNC_B128(g + 8, l + 8);
#else
    const uint4* s4 = (const uint4*)g;
    uint4* d4 = (uint4*)l;
    d4[0] = s4[0];
    d4[1] = s4[1];
    __builtin_prefetch(g + strideElems, 0, 0);  // global_prefetch_b8 hint
#endif
}

// A fragment: 16x32 bf16, M-major LDS tile sW[128][32].
// Lane layout (ISA 7.12.2): lanes 0-15 row M=lane, K={0..7,16..23};
// lanes 16-31 row M=lane-16, K={8..15,24..31}. Two contiguous 8xbf16 runs.
__device__ inline Frag load_a(const __bf16* sW, int row, int half) {
    Frag f;
    const uint4* p = (const uint4*)(sW + row * 32);
    f.u[0] = p[half];       // K = half*8 .. half*8+7
    f.u[1] = p[2 + half];   // K = 16+half*8 .. 16+half*8+7
    return f;
}

// B fragment: 32x16 bf16, column-major LDS tile sS[128 n][32 k].
// Lane layout: lanes 0-15 col N=lane, K={0..7,16..23}; lanes 16-31 col N=lane-16, K={8..15,24..31}.
__device__ inline Frag load_b(const __bf16* sS, int col, int half) {
    Frag f;
    const uint4* p = (const uint4*)(sS + col * 32);
    f.u[0] = p[half];
    f.u[1] = p[2 + half];
    return f;
}

__device__ inline v8f wmma_bf16(const Frag& a, const Frag& b, v8f c) {
    return __builtin_amdgcn_wmma_f32_16x16x32_bf16(
        /*neg_a=*/false, a.v, /*neg_b=*/false, b.v,
        /*c_mod=*/(short)0, c, /*reuse_a=*/false, /*reuse_b=*/false);
}

// ---------------------------------------------------------------------------
// Problem constants: B=4, C=O=256, H=W=64, K=9
// ---------------------------------------------------------------------------
#define NB   4
#define NC   256
#define HH   64
#define WW   64
#define NPIX 4096           // H*W
#define CK   2304           // C*9
#define KD   1024           // deconv GEMM K = 256*4

// ---------------------------------------------------------------------------
// fp32 -> bf16 flat convert
// ---------------------------------------------------------------------------
__global__ void cvt_bf16_kernel(const float* __restrict__ in,
                                __bf16* __restrict__ out, int n) {
    int i = blockIdx.x * blockDim.x + threadIdx.x;
    if (i < n) out[i] = (__bf16)in[i];
}

// ---------------------------------------------------------------------------
// Repack deconv weight into 4 parity classes:
// out[cls][o2][o*4 + ai*2+bi] = w_deconv[o][o2][3-(py+2ai)][3-(px+2bi)]
// ---------------------------------------------------------------------------
__global__ void pack_wdeconv_kernel(const float* __restrict__ wd,
                                    __bf16* __restrict__ out) {
    int idx = blockIdx.x * blockDim.x + threadIdx.x;
    if (idx >= 4 * 256 * 256 * 4) return;
    int t   = idx & 3;
    int o   = (idx >> 2) & 255;
    int o2  = (idx >> 10) & 255;
    int cls = idx >> 18;
    int py = cls >> 1, px = cls & 1;
    int ai = t >> 1,  bi = t & 1;
    int a = py + 2 * ai;
    int bq = px + 2 * bi;
    out[idx] = (__bf16)wd[((o * 256 + o2) * 4 + (3 - a)) * 4 + (3 - bq)];
}

// ---------------------------------------------------------------------------
// Offset conv (256->27ch, 3x3, fp32 direct) fused with sigmoid + bilinear
// coefficient precompute. One thread per (b, k, pixel).
// ---------------------------------------------------------------------------
__global__ void coeffs_kernel(const float* __restrict__ x,
                              const float* __restrict__ w_off,
                              const float* __restrict__ b_off,
                              int2* __restrict__ cYX,
                              float4* __restrict__ cW) {
    int idx = blockIdx.x * blockDim.x + threadIdx.x;
    if (idx >= NB * 9 * NPIX) return;
    int p = idx & (NPIX - 1);
    int k = (idx >> 12) % 9;
    int b = idx / (9 * NPIX);
    int h = p >> 6, w = p & 63;

    int chy = 2 * k, chx = 2 * k + 1, chm = 18 + k;
    float ady = b_off[chy], adx = b_off[chx], am = b_off[chm];
    const float* xb = x + (size_t)b * NC * NPIX;
    for (int c = 0; c < NC; ++c) {
        const float* xc = xb + c * NPIX;
        const float* wy = w_off + (chy * NC + c) * 9;
        const float* wx = w_off + (chx * NC + c) * 9;
        const float* wm = w_off + (chm * NC + c) * 9;
#pragma unroll
        for (int t = 0; t < 9; ++t) {
            int yy = h + t / 3 - 1;
            int xx = w + t % 3 - 1;
            float xv = (yy >= 0 && yy < HH && xx >= 0 && xx < WW)
                           ? xc[yy * WW + xx] : 0.f;
            ady += xv * wy[t];
            adx += xv * wx[t];
            am  += xv * wm[t];
        }
    }
    float m  = 1.f / (1.f + __expf(-am));
    float ky = (float)(k / 3 - 1), kx = (float)(k % 3 - 1);
    float py = ady + (float)h + ky;
    float px = adx + (float)w + kx;
    float y0 = floorf(py), x0 = floorf(px);
    float ly = py - y0,   lx = px - x0;
    cYX[idx] = make_int2((int)y0, (int)x0);
    cW[idx]  = make_float4((1.f - ly) * (1.f - lx) * m,
                           (1.f - ly) * lx * m,
                           ly * (1.f - lx) * m,
                           ly * lx * m);
}

// ---------------------------------------------------------------------------
// Deformable-conv GEMM: y[o,p] = sum_ck w2[o,ck] * sampled[ck,p].
// W tile double-buffered in LDS via async global->LDS copies (ASYNCcnt);
// B tile = fused bilinear gather of L2-resident bf16 x.
// Tile: M=128 (8 waves x 16), N=128 pixels, K-step 32. 256 threads.
// ---------------------------------------------------------------------------
__global__ __launch_bounds__(256) void dcn_gemm_kernel(
    const __bf16* __restrict__ xb, const __bf16* __restrict__ w2,
    const int2* __restrict__ cYX, const float4* __restrict__ cW,
    const float* __restrict__ b_dcn, const float* __restrict__ gamma,
    const float* __restrict__ beta, const float* __restrict__ mean,
    const float* __restrict__ var, __bf16* __restrict__ y) {
    __shared__ __align__(16) __bf16 sW[2][128 * 32];   // [buf][m][k]
    __shared__ __align__(16) __bf16 sS[128 * 32];      // [n][k]
    __shared__ int2   sC[9 * 128];
    __shared__ float4 sCW[9 * 128];

    const int nt = blockIdx.x;          // 0..31 pixel tile
    const int mt = blockIdx.y;          // 0..1  output-channel tile
    const int b  = blockIdx.z;          // 0..3  batch
    const int pixBase = nt * 128;
    const int mBase   = mt * 128;
    const int tid  = threadIdx.x;
    const int lane = tid & 31;
    const int wave = tid >> 5;
    const int wrow = tid >> 1, whalf = tid & 1;

    const __bf16* wtile = w2 + (size_t)mBase * CK;

    // Kick off async copy of the first W tile, overlap with coeff staging.
    stage_w_tile(wtile, sW[0], wrow, whalf, CK);

    for (int e = tid; e < 9 * 128; e += 256) {
        int k = e >> 7, n = e & 127;
        int ci = (b * 9 + k) * NPIX + pixBase + n;
        sC[e]  = cYX[ci];
        sCW[e] = cW[ci];
    }
    __syncthreads();

    v8f acc[8];
#pragma unroll
    for (int i = 0; i < 8; ++i) acc[i] = zero8();

    const __bf16* xc = xb + (size_t)b * NC * NPIX;
    const int nS = tid & 127, kset = tid >> 7;

    int buf = 0;
    for (int ck0 = 0; ck0 < CK; ck0 += 32, buf ^= 1) {
        // -- B tile: fused bilinear gather of bf16 x (L2-resident)
#pragma unroll 4
        for (int kr = kset * 16; kr < kset * 16 + 16; ++kr) {
            int ckk = ck0 + kr;
            int c = ckk / 9;
            int k = ckk - c * 9;
            int2   yx = sC[k * 128 + nS];
            float4 wv = sCW[k * 128 + nS];
            const __bf16* xcc = xc + c * NPIX;
            int y0 = yx.x, x0 = yx.y;
            bool yv0 = (y0 >= 0) & (y0 < HH);
            bool yv1 = (y0 + 1 >= 0) & (y0 + 1 < HH);
            bool xv0 = (x0 >= 0) & (x0 < WW);
            bool xv1 = (x0 + 1 >= 0) & (x0 + 1 < WW);
            float v = 0.f;
            if (yv0 & xv0) v += wv.x * (float)xcc[y0 * WW + x0];
            if (yv0 & xv1) v += wv.y * (float)xcc[y0 * WW + x0 + 1];
            if (yv1 & xv0) v += wv.z * (float)xcc[(y0 + 1) * WW + x0];
            if (yv1 & xv1) v += wv.w * (float)xcc[(y0 + 1) * WW + x0 + 1];
            sS[nS * 32 + kr] = (__bf16)v;
        }

        wait_async0();         // W tile `buf` resident (this wave's copies)
        __syncthreads();       // publish W tile + S tile to all waves

        // Issue async copy of the next W tile; overlaps the WMMAs below and
        // the next iteration's gather phase.
        if (ck0 + 32 < CK)
            stage_w_tile(wtile + ck0 + 32, sW[buf ^ 1], wrow, whalf, CK);

        Frag a = load_a(sW[buf], wave * 16 + (lane & 15), lane >> 4);
#pragma unroll
        for (int ntile = 0; ntile < 8; ++ntile) {
            Frag bf = load_b(sS, ntile * 16 + (lane & 15), lane >> 4);
            acc[ntile] = wmma_bf16(a, bf, acc[ntile]);
        }
        __syncthreads();       // guard sS against next iteration's stores
    }

    // Epilogue: C/D layout lane n = lane&15, M = (lane>>4)*8 + j.
    const int nloc = lane & 15, half = lane >> 4;
#pragma unroll
    for (int j = 0; j < 8; ++j) {
        int o = mBase + wave * 16 + half * 8 + j;
        float inv = gamma[o] * rsqrtf(var[o] + 1e-5f);
        float add = beta[o] - mean[o] * inv + b_dcn[o] * inv;
#pragma unroll
        for (int t = 0; t < 8; ++t) {
            float v = acc[t][j] * inv + add;
            v = v > 0.f ? v : 0.f;
            y[((size_t)b * NC + o) * NPIX + pixBase + t * 16 + nloc] = (__bf16)v;
        }
    }
}

// ---------------------------------------------------------------------------
// Transposed-conv 4x4/s2 as 4 parity-class GEMMs:
// out[b,o2,2P+py,2Q+px] = sum_{o,ai,bi} wcls[o2][o*4+ai*2+bi]*y[b,o,P+1-ai,Q+1-bi]
// ---------------------------------------------------------------------------
__global__ __launch_bounds__(256) void deconv_gemm_kernel(
    const __bf16* __restrict__ y, const __bf16* __restrict__ wd,
    float* __restrict__ out) {
    __shared__ __align__(16) __bf16 sW[2][128 * 32];
    __shared__ __align__(16) __bf16 sS[128 * 32];

    const int nt = blockIdx.x;          // 0..31
    const int mt = blockIdx.y;          // 0..1
    const int bc = blockIdx.z;          // 0..15 = b*4 + cls
    const int b = bc >> 2, cls = bc & 3;
    const int py = cls >> 1, px = cls & 1;
    const int pixBase = nt * 128, mBase = mt * 128;
    const int tid = threadIdx.x, lane = tid & 31, wave = tid >> 5;
    const int wrow = tid >> 1, whalf = tid & 1;

    const __bf16* wtile = wd + (size_t)cls * 256 * KD + (size_t)mBase * KD;
    const __bf16* yb    = y + (size_t)b * NC * NPIX;

    stage_w_tile(wtile, sW[0], wrow, whalf, KD);

    v8f acc[8];
#pragma unroll
    for (int i = 0; i < 8; ++i) acc[i] = zero8();

    const int nS = tid & 127, kset = tid >> 7;
    const int pix = pixBase + nS;
    const int P = pix >> 6, Q = pix & 63;

    int buf = 0;
    for (int ck0 = 0; ck0 < KD; ck0 += 32, buf ^= 1) {
#pragma unroll 8
        for (int kr = kset * 16; kr < kset * 16 + 16; ++kr) {
            int ko = ck0 + kr;
            int o = ko >> 2, t = ko & 3;
            int ai = t >> 1, bi = t & 1;
            int yy = P + 1 - ai;
            int xx = Q + 1 - bi;
            __bf16 v = (__bf16)0.f;
            if ((yy < HH) & (xx < WW)) v = yb[o * NPIX + yy * WW + xx];
            sS[nS * 32 + kr] = v;
        }

        wait_async0();
        __syncthreads();

        if (ck0 + 32 < KD)
            stage_w_tile(wtile + ck0 + 32, sW[buf ^ 1], wrow, whalf, KD);

        Frag a = load_a(sW[buf], wave * 16 + (lane & 15), lane >> 4);
#pragma unroll
        for (int ntile = 0; ntile < 8; ++ntile) {
            Frag bf = load_b(sS, ntile * 16 + (lane & 15), lane >> 4);
            acc[ntile] = wmma_bf16(a, bf, acc[ntile]);
        }
        __syncthreads();
    }

    const int nloc = lane & 15, half = lane >> 4;
#pragma unroll
    for (int j = 0; j < 8; ++j) {
        int o2 = mBase + wave * 16 + half * 8 + j;
#pragma unroll
        for (int t = 0; t < 8; ++t) {
            int pp = pixBase + t * 16 + nloc;
            int Po = pp >> 6, Qo = pp & 63;
            out[(((size_t)b * 256 + o2) * 128 + (2 * Po + py)) * 128 +
                (2 * Qo + px)] = acc[t][j];
        }
    }
}

// ---------------------------------------------------------------------------
// Host side
// ---------------------------------------------------------------------------
extern "C" void kernel_launch(void* const* d_in, const int* in_sizes, int n_in,
                              void* d_out, int out_size, void* d_ws, size_t ws_size,
                              hipStream_t stream) {
    const float* x        = (const float*)d_in[0];
    const float* w_off    = (const float*)d_in[1];
    const float* b_off    = (const float*)d_in[2];
    const float* w_dcn    = (const float*)d_in[3];
    const float* b_dcn    = (const float*)d_in[4];
    const float* bn_gamma = (const float*)d_in[5];
    const float* bn_beta  = (const float*)d_in[6];
    const float* bn_mean  = (const float*)d_in[7];
    const float* bn_var   = (const float*)d_in[8];
    const float* w_deconv = (const float*)d_in[9];
    float* out = (float*)d_out;

    char* ws = (char*)d_ws;
    // Workspace layout (all 256B aligned), ~23.6 MB total:
    int2*   cYX = (int2*)  (ws + 0);                      //  1,179,648 B
    float4* cW  = (float4*)(ws + 1179648);                //  2,359,296 B
    __bf16* xb  = (__bf16*)(ws + 3538944);                //  8,388,608 B
    __bf16* w2  = (__bf16*)(ws + 11927552);               //  1,179,648 B
    __bf16* yb  = (__bf16*)(ws + 13107200);               //  8,388,608 B
    __bf16* wdp = (__bf16*)(ws + 21495808);               //  2,097,152 B

    const int nX  = NB * NC * NPIX;     // 4,194,304
    const int nW2 = NC * CK;            //   589,824
    const int nWD = 4 * 256 * 256 * 4;  // 1,048,576

    cvt_bf16_kernel<<<(nX + 255) / 256, 256, 0, stream>>>(x, xb, nX);
    cvt_bf16_kernel<<<(nW2 + 255) / 256, 256, 0, stream>>>(w_dcn, w2, nW2);
    pack_wdeconv_kernel<<<(nWD + 255) / 256, 256, 0, stream>>>(w_deconv, wdp);

    const int nCoef = NB * 9 * NPIX;    // 147,456
    coeffs_kernel<<<(nCoef + 127) / 128, 128, 0, stream>>>(x, w_off, b_off, cYX, cW);

    dcn_gemm_kernel<<<dim3(32, 2, NB), 256, 0, stream>>>(
        xb, w2, cYX, cW, b_dcn, bn_gamma, bn_beta, bn_mean, bn_var, yb);

    deconv_gemm_kernel<<<dim3(32, 2, NB * 4), 256, 0, stream>>>(yb, wdp, out);
}
